// FreqAttentionLayer_34497177322062
// MI455X (gfx1250) — compile-verified, matrix operand order checked by reading, not run on hardware
//
#include <hip/hip_runtime.h>

typedef __attribute__((ext_vector_type(8)))  float  v8f;
typedef __attribute__((ext_vector_type(8)))  __bf16 v8bf;
typedef __attribute__((ext_vector_type(16))) __bf16 v16bf;

#define B_     2
#define L_     2048
#define H_     8
#define DK_    64
#define DIN_   1024
#define DPROJ_ 1024
#define M_     (B_ * L_)   // 4096 rows for every GEMM

#define LDS_STRIDE 40      // 32 K-elements + 8 pad: 80B rows, 16B aligned

// ---------------------------------------------------------------- helpers ---
__device__ __forceinline__ __bf16 to_bf16(float f) {
  union { float f; unsigned u; } x; x.f = f;
  unsigned r = x.u + 0x7FFFu + ((x.u >> 16) & 1u);   // round-to-nearest-even
  union { unsigned short s; __bf16 b; } y;
  y.s = (unsigned short)(r >> 16);
  return y.b;
}

__global__ void cvt_f32_bf16(const float* __restrict__ src,
                             __bf16* __restrict__ dst, int n) {
  int i = blockIdx.x * blockDim.x + threadIdx.x;
  if (i < n) dst[i] = to_bf16(src[i]);
}

// Load one 16x32 bf16 WMMA fragment from an LDS tile row.
// halves 0..7 = K kb..kb+7, halves 8..15 = K kb+16..kb+23 (ISA 7.12.2 layout).
__device__ __forceinline__ v16bf load_frag_lds(const __bf16* p) {
  v8bf lo = *reinterpret_cast<const v8bf*>(p);
  v8bf hi = *reinterpret_cast<const v8bf*>(p + 16);
  return __builtin_shufflevector(lo, hi,
      0, 1, 2, 3, 4, 5, 6, 7, 8, 9, 10, 11, 12, 13, 14, 15);
}

// ------------------------------------------------------------------- GEMM ---
// C(MxN, f32) = A(MxK, bf16) @ Bw^T + bias,  Bw stored as N rows x K cols.
// Block = 8 waves / 128x128 tile; async global->LDS staging per 32-K step;
// each wave owns a 64x32 block (4x2 WMMA tiles).
__global__ __launch_bounds__(256) void gemm_bf16_wmma(
    const __bf16* __restrict__ A, const __bf16* __restrict__ Bw,
    const float* __restrict__ bias, float* __restrict__ C,
    int M, int N, int K)
{
  __shared__ __bf16 shA[128 * LDS_STRIDE];
  __shared__ __bf16 shB[128 * LDS_STRIDE];

  const int tid  = threadIdx.x;
  const int lane = tid & 31;
  const int wave = tid >> 5;
  const int wm   = wave >> 2;                 // 0..1  -> 64-row slice
  const int wn   = wave & 3;                  // 0..3  -> 32-col slice
  const int m0   = blockIdx.x * 128;
  const int n0   = blockIdx.y * 128;
  const int r    = lane & 15;                 // fragment row
  const int kb   = (lane >> 4) * 8;           // K sub-offset within fragment

  // LDS byte offsets (generic shared pointer low 32 bits == LDS offset)
  const unsigned ldsA = (unsigned)(unsigned long long)(void*)shA;
  const unsigned ldsB = (unsigned)(unsigned long long)(void*)shB;

  v8f acc[4][2] = {};

  for (int k0 = 0; k0 < K; k0 += 32) {
    // ---- async stage 128x32 A-tile and 128x32 B-tile into LDS ----
    // tile = 512 x 16B chunks; 2 chunks per thread per tile
#pragma unroll
    for (int t = 0; t < 2; ++t) {
      const int c   = tid + 256 * t;
      const int row = c >> 2;                 // 0..127
      const int col = (c & 3) * 8;            // 0,8,16,24 (bf16 elements)
      const unsigned long long ga =
          (unsigned long long)(A + (size_t)(m0 + row) * K + k0 + col);
      const unsigned long long gb =
          (unsigned long long)(Bw + (size_t)(n0 + row) * K + k0 + col);
      const unsigned la = ldsA + (unsigned)(row * LDS_STRIDE + col) * 2u;
      const unsigned lb = ldsB + (unsigned)(row * LDS_STRIDE + col) * 2u;
      asm volatile("global_load_async_to_lds_b128 %0, %1, off"
                   :: "v"(la), "v"(ga) : "memory");
      asm volatile("global_load_async_to_lds_b128 %0, %1, off"
                   :: "v"(lb), "v"(gb) : "memory");
    }
    asm volatile("s_wait_asynccnt 0x0" ::: "memory");
    __syncthreads();

    // ---- fragments from LDS + 8 WMMAs ----
    v16bf af[4], bfr[2];
#pragma unroll
    for (int i = 0; i < 4; ++i)
      af[i] = load_frag_lds(shA + (wm * 64 + 16 * i + r) * LDS_STRIDE + kb);
#pragma unroll
    for (int j = 0; j < 2; ++j)
      bfr[j] = load_frag_lds(shB + (wn * 32 + 16 * j + r) * LDS_STRIDE + kb);
#pragma unroll
    for (int i = 0; i < 4; ++i)
#pragma unroll
      for (int j = 0; j < 2; ++j)
        acc[i][j] = __builtin_amdgcn_wmma_f32_16x16x32_bf16(
            false, af[i], false, bfr[j], (short)0, acc[i][j], false, false);
    __syncthreads();
  }

  // C/D layout: lane l -> col = l&15, rows = e + 8*(l>>4)
  const int rhi = (lane >> 4) * 8;
  const int cn  = lane & 15;
  const int mw  = m0 + wm * 64;
  const int nw  = n0 + wn * 32;
#pragma unroll
  for (int i = 0; i < 4; ++i) {
#pragma unroll
    for (int j = 0; j < 2; ++j) {
      const int col = nw + 16 * j + cn;
      const float bc = bias[col];
#pragma unroll
      for (int e = 0; e < 8; ++e) {
        const int row = mw + 16 * i + rhi + e;
        C[(size_t)row * N + col] = acc[i][j][e] + bc;
      }
    }
  }
}

// -------------------------------------------------------------- attention ---
// One 64-thread block per (b, n, h): 64x64 scores over the d_keys axis with
// inner dim E=2, online softmax, bf16 output feeding the final GEMM.
__global__ __launch_bounds__(64) void freq_attn(
    const float* __restrict__ qp, const float* __restrict__ kp,
    const float* __restrict__ vp, __bf16* __restrict__ outp)
{
  __shared__ float ks0[DK_], ks1[DK_], vs0[DK_], vs1[DK_];
  const int idx = blockIdx.x;
  const int h   = idx & (H_ - 1);
  const int bn  = idx >> 3;
  const size_t base = (size_t)bn * DPROJ_;
  const int t   = threadIdx.x;          // row l, also loader index s
  const int off = t * (H_ * 2) + h * 2; // p = dk*16 + h*2 + e

  ks0[t] = kp[base + off];  ks1[t] = kp[base + off + 1];
  vs0[t] = vp[base + off];  vs1[t] = vp[base + off + 1];
  __syncthreads();

  const float q0 = qp[base + off], q1 = qp[base + off + 1];
  const float scale = 0.70710678118654752f;   // 1/sqrt(E), E=2

  float mx = -3.0e38f, ssum = 0.f, o0 = 0.f, o1 = 0.f;
#pragma unroll 8
  for (int s = 0; s < DK_; ++s) {
    float sc   = (q0 * ks0[s] + q1 * ks1[s]) * scale;
    float mn   = fmaxf(mx, sc);
    float corr = __expf(mx - mn);
    float p    = __expf(sc - mn);
    ssum = ssum * corr + p;
    o0   = o0 * corr + p * vs0[s];
    o1   = o1 * corr + p * vs1[s];
    mx   = mn;
  }
  const float inv = 1.0f / ssum;
  outp[base + off]     = to_bf16(o0 * inv);
  outp[base + off + 1] = to_bf16(o1 * inv);
}

// ------------------------------------------------------------------- host ---
extern "C" void kernel_launch(void* const* d_in, const int* in_sizes, int n_in,
                              void* d_out, int out_size, void* d_ws, size_t ws_size,
                              hipStream_t stream) {
  (void)in_sizes; (void)n_in; (void)out_size; (void)ws_size;
  const float* q  = (const float*)d_in[0];
  const float* k  = (const float*)d_in[1];
  const float* v  = (const float*)d_in[2];
  const float* Wq = (const float*)d_in[3];
  const float* bq = (const float*)d_in[4];
  const float* Wk = (const float*)d_in[5];
  const float* bk = (const float*)d_in[6];
  const float* Wv = (const float*)d_in[7];
  const float* bv = (const float*)d_in[8];
  const float* Wo = (const float*)d_in[9];
  const float* bo = (const float*)d_in[10];
  float* out = (float*)d_out;

  const size_t MK = (size_t)M_ * DIN_;        // 4096*1024
  const size_t NK = (size_t)DPROJ_ * DIN_;    // 1024*1024

  char* ws = (char*)d_ws;
  __bf16* qb    = (__bf16*)ws; ws += MK * 2;
  __bf16* kb2   = (__bf16*)ws; ws += MK * 2;
  __bf16* vb    = (__bf16*)ws; ws += MK * 2;
  __bf16* Wqb   = (__bf16*)ws; ws += NK * 2;
  __bf16* Wkb   = (__bf16*)ws; ws += NK * 2;
  __bf16* Wvb   = (__bf16*)ws; ws += NK * 2;
  __bf16* Wob   = (__bf16*)ws; ws += NK * 2;
  float*  qproj = (float*)ws;  ws += MK * 4;
  float*  kproj = (float*)ws;  ws += MK * 4;
  float*  vproj = (float*)ws;  ws += MK * 4;
  __bf16* attnb = (__bf16*)ws; ws += MK * 2;

  const int CT = 256;
  cvt_f32_bf16<<<(int)((MK + CT - 1) / CT), CT, 0, stream>>>(q,  qb,  (int)MK);
  cvt_f32_bf16<<<(int)((MK + CT - 1) / CT), CT, 0, stream>>>(k,  kb2, (int)MK);
  cvt_f32_bf16<<<(int)((MK + CT - 1) / CT), CT, 0, stream>>>(v,  vb,  (int)MK);
  cvt_f32_bf16<<<(int)((NK + CT - 1) / CT), CT, 0, stream>>>(Wq, Wqb, (int)NK);
  cvt_f32_bf16<<<(int)((NK + CT - 1) / CT), CT, 0, stream>>>(Wk, Wkb, (int)NK);
  cvt_f32_bf16<<<(int)((NK + CT - 1) / CT), CT, 0, stream>>>(Wv, Wvb, (int)NK);
  cvt_f32_bf16<<<(int)((NK + CT - 1) / CT), CT, 0, stream>>>(Wo, Wob, (int)NK);

  dim3 g(M_ / 128, DPROJ_ / 128);   // 32 x 8
  gemm_bf16_wmma<<<g, 256, 0, stream>>>(qb,  Wqb, bq, qproj, M_, DPROJ_, DIN_);
  gemm_bf16_wmma<<<g, 256, 0, stream>>>(kb2, Wkb, bk, kproj, M_, DPROJ_, DIN_);
  gemm_bf16_wmma<<<g, 256, 0, stream>>>(vb,  Wvb, bv, vproj, M_, DPROJ_, DIN_);

  freq_attn<<<B_ * L_ * H_, 64, 0, stream>>>(qproj, kproj, vproj, attnb);

  gemm_bf16_wmma<<<g, 256, 0, stream>>>(attnb, Wob, bo, out, M_, DIN_, DPROJ_);
}
